// PolyMatchingLoss_42262478192860
// MI455X (gfx1250) — compile-verified, matrix operand order checked by reading, not run on hardware
//
#include <hip/hip_runtime.h>
#include <hip/hip_bf16.h>

typedef __attribute__((ext_vector_type(16))) _Float16 v16h;
typedef __attribute__((ext_vector_type(8)))  float    v8f;

#define PNUM  512
#define BATCH 256
#define LPRED 2

__device__ __forceinline__ float smooth_l1(float x) {
    float ax = __builtin_fabsf(x);
    return ax < 1.0f ? 0.5f * ax * ax : ax - 0.5f;
}

// One workgroup per (pred_idx, batch). 256 threads = 8 wave32s.
// Each wave handles 4 tiles of 16 r-values; the p-reduction (512 terms)
// runs on the matrix unit: A = f16 cost tile (16r x 32p), B = ones,
// C/D = f32 accumulators -> 16 row sums per v_wmma_f32_16x16x32_f16.
// Two independent accumulator chains hide WMMA latency under the VALU
// smooth-L1 packing of the other chain.
__global__ void __launch_bounds__(256)
poly_match_min_kernel(const float* __restrict__ pred_list,
                      const float* __restrict__ gt,
                      float* __restrict__ block_min) {
    __shared__ float s_px[PNUM];
    __shared__ float s_py[PNUM];
    __shared__ float s_gx[PNUM];
    __shared__ float s_gy[PNUM];
    __shared__ float s_wmin[8];

    const int tid  = threadIdx.x;
    const int blk  = blockIdx.x;           // blk = i*BATCH + b
    const int b    = blk & (BATCH - 1);

    // Stage pred[i,b] and gt[b] into LDS (each 512 x float2 = 4KB).
    const float2* __restrict__ pred2 = (const float2*)(pred_list) + (size_t)blk * PNUM;
    const float2* __restrict__ gt2   = (const float2*)(gt)        + (size_t)b   * PNUM;
    for (int p = tid; p < PNUM; p += 256) {
        float2 pv = pred2[p];
        float2 gv = gt2[p];
        s_px[p] = pv.x;  s_py[p] = pv.y;
        s_gx[p] = gv.x;  s_gy[p] = gv.y;
    }
    __syncthreads();

    const int wave = tid >> 5;             // 0..7
    const int lane = tid & 31;             // 0..31
    const int rcol = lane & 15;            // M index within 16-wide tile
    const int koff = (lane < 16) ? 0 : 8;  // A-matrix 16-bit layout half-K offset

    // B = all-ones f16 matrix: with ones, D = per-row sums of A regardless
    // of B's lane layout.
    v16h ones;
#pragma unroll
    for (int j = 0; j < 16; ++j) ones[j] = (_Float16)1.0f;

    // Elementwise running min of row-sums across tiles; cross-lane
    // reduction deferred to a single exchange at the end.
    v8f runmin;
#pragma unroll
    for (int j = 0; j < 8; ++j) runmin[j] = 3.0e38f;

    for (int t = 0; t < 4; ++t) {
        const int rbase = wave * 64 + t * 16;
        const int r     = rbase + rcol;
        v8f acc0 = {};
        v8f acc1 = {};

        for (int pbase = 0; pbase < PNUM; pbase += 64) {
            v16h a0, a1;
#pragma unroll
            for (int j = 0; j < 16; ++j) {
                // 16-bit A 16x32 layout: lanes 0-15 hold K={0..7,16..23},
                // lanes 16-31 hold K={8..15,24..31}; v16h elem 2v,2v+1 are
                // VGPR v's [15:0],[31:16].
                const int k  = ((j < 8) ? j : j + 8) + koff;
                const int p0 = pbase + k;
                const int p1 = p0 + 32;
                const int g0 = (r + p0) & (PNUM - 1);
                const int g1 = (r + p1) & (PNUM - 1);
                float c0 = smooth_l1(s_px[p0] - s_gx[g0]) +
                           smooth_l1(s_py[p0] - s_gy[g0]);
                float c1 = smooth_l1(s_px[p1] - s_gx[g1]) +
                           smooth_l1(s_py[p1] - s_gy[g1]);
                a0[j] = (_Float16)c0;
                a1[j] = (_Float16)c1;
            }
            // Two independent D = A x ones + C chains.
            acc0 = __builtin_amdgcn_wmma_f32_16x16x32_f16(
                       false, a0, false, ones, (short)0, acc0, false, false);
            acc1 = __builtin_amdgcn_wmma_f32_16x16x32_f16(
                       false, a1, false, ones, (short)0, acc1, false, false);
        }

#pragma unroll
        for (int j = 0; j < 8; ++j)
            runmin[j] = fminf(runmin[j], acc0[j] + acc1[j]);
    }

    // Lane holds 8 row-sum minima (M = j + 8*(lane/16)); values replicated
    // across the 16 N-columns, so one xor-16 exchange covers all 16 rows.
    float m = runmin[0];
#pragma unroll
    for (int j = 1; j < 8; ++j) m = fminf(m, runmin[j]);
    m = fminf(m, __shfl_xor(m, 16, 32));

    if (lane == 0) s_wmin[wave] = m;
    __syncthreads();

    if (tid == 0) {
        float bm = s_wmin[0];
#pragma unroll
        for (int w = 1; w < 8; ++w) bm = fminf(bm, s_wmin[w]);
        block_min[blk] = bm;               // = min_r Σ_p cost  (un-normalized)
    }
}

// Deterministic tree reduction of the 512 per-block minima.
// loss = Σ block_min / (PNUM * LPRED * BATCH) = Σ / (512*512)
__global__ void __launch_bounds__(256)
poly_match_reduce_kernel(const float* __restrict__ block_min,
                         float* __restrict__ out) {
    __shared__ float s[256];
    const int t = threadIdx.x;
    s[t] = block_min[t] + block_min[t + 256];
    __syncthreads();
    for (int stride = 128; stride > 0; stride >>= 1) {
        if (t < stride) s[t] += s[t + stride];
        __syncthreads();
    }
    if (t == 0) out[0] = s[0] * (1.0f / (512.0f * 512.0f));
}

extern "C" void kernel_launch(void* const* d_in, const int* in_sizes, int n_in,
                              void* d_out, int out_size, void* d_ws, size_t ws_size,
                              hipStream_t stream) {
    (void)in_sizes; (void)n_in; (void)out_size; (void)ws_size;
    const float* pred_list = (const float*)d_in[0];   // (2, 256, 512, 2) f32
    const float* gt        = (const float*)d_in[1];   // (256, 512, 2) f32
    float* out             = (float*)d_out;           // scalar f32
    float* block_min       = (float*)d_ws;            // 512 floats of scratch

    poly_match_min_kernel<<<LPRED * BATCH, 256, 0, stream>>>(pred_list, gt, block_min);
    poly_match_reduce_kernel<<<1, 256, 0, stream>>>(block_min, out);
}
